// GeneralizedCrossMessage_81363860456152
// MI455X (gfx1250) — matrix operand order repaired and checked by reading, not compiled, not run on hardware
//
#include <hip/hip_runtime.h>

// ---------------- types ----------------
typedef __attribute__((ext_vector_type(16))) __bf16 v16bf;
typedef __attribute__((ext_vector_type(8)))  __bf16 v8bf;
typedef __attribute__((ext_vector_type(8)))  float  v8f;
typedef __attribute__((ext_vector_type(4)))  float  v4f;

// ---------------- problem constants ----------------
static constexpr int Bc  = 4;
static constexpr int Sc  = 512;
static constexpr int Hc  = 768;
static constexpr int Dc  = 96;     // projection dim
static constexpr int MHc = 96;     // score MLP hidden
static constexpr int K1c = 3 * Hc; // 2304, value MLP input
static constexpr int OHc = 768;    // OUT_HID
static constexpr float NEGMAX = -3.4028234663852886e38f;

__device__ __forceinline__ v8f wmma_bf16(v16bf a, v16bf b, v8f c) {
  // (neg_a, A, neg_b, B, c_mod, C, reuse_a, reuse_b)
  return __builtin_amdgcn_wmma_f32_16x16x32_bf16(false, a, false, b, (short)0, c,
                                                 false, false);
}

// ---------------- weight transpose + bf16 convert: in[K][N] -> out[N][K] ----
__global__ void transpose_bf16_kernel(const float* __restrict__ in,
                                      __bf16* __restrict__ out, int K, int N) {
  int idx = blockIdx.x * 256 + threadIdx.x;
  if (idx < K * N) {
    int k = idx / N, n = idx % N;
    out[(size_t)n * K + k] = (__bf16)in[idx];
  }
}

// ---------------- projections: Zj = H_j@Wpj, Zi = H_i@Wpi ----------------
__global__ void proj_kernel(const float* __restrict__ Hj,
                            const float* __restrict__ Hi,
                            const float* __restrict__ Wpj,
                            const float* __restrict__ Wpi,
                            float* __restrict__ Zj, float* __restrict__ Zi) {
  int row = blockIdx.x;                 // b*S + s
  int t = threadIdx.x;
  const float* hj = Hj + (size_t)row * Hc;
  const float* hi = Hi + (size_t)row * Hc;
  if (t < Dc) {
    float a = 0.f;
    for (int d = 0; d < Hc; ++d) a += hj[d] * Wpj[(size_t)d * Dc + t];
    Zj[(size_t)row * Dc + t] = a;
  } else if (t < 2 * Dc) {
    int c = t - Dc;
    float a = 0.f;
    for (int d = 0; d < Hc; ++d) a += hi[d] * Wpi[(size_t)d * Dc + c];
    Zi[(size_t)row * Dc + c] = a;
  }
}

// ------- separable score terms: Aj = Zj@Ws1[0:96], Bi = Zi@Ws1[96:192] -----
__global__ void ajbi_kernel(const float* __restrict__ Zj,
                            const float* __restrict__ Zi,
                            const float* __restrict__ Ws1,
                            float* __restrict__ Aj, float* __restrict__ Bi) {
  int row = blockIdx.x;
  int t = threadIdx.x;
  if (t < MHc) {
    const float* z = Zj + (size_t)row * Dc;
    float a = 0.f;
    for (int d = 0; d < Dc; ++d) a += z[d] * Ws1[(size_t)d * MHc + t];
    Aj[(size_t)row * MHc + t] = a;
  } else if (t < 2 * MHc) {
    int m = t - MHc;
    const float* z = Zi + (size_t)row * Dc;
    float a = 0.f;
    for (int d = 0; d < Dc; ++d) a += z[d] * Ws1[(size_t)(Dc + d) * MHc + m];
    Bi[(size_t)row * MHc + m] = a;
  }
}

// ---------------- heavy pair-score kernel (WMMA bf16) ----------------
// Grid (B*S, 4); block = 8 waves; wave w of block (bj,q) owns exactly ONE
// i-tile of 16 keys: itile = q*8 + w.  No outer tile loop -> no loop-invariant
// B-fragments for LICM to hoist/spill.
// Per tile: hidden[i,m] = relu(F @ Ws1t + Aj[j] + Bi[i] + bs1),
//           F[i,k] = (k<96) ? Zj[j,k]*Zi[i,k] : |Zj[j,k-96]-Zi[i,k-96]|
// kr loop kept rolled: per iteration one Zj/Zi load set feeds the product
// (kt=kr) and abs-diff (kt=kr+3) fragments and 12 WMMAs.
__global__ __launch_bounds__(256, 4) void pair_kernel(
    const float* __restrict__ Zj, const float* __restrict__ Zi,
    const __bf16* __restrict__ Ws1t,   // [96][192] (rows 192..383 of Ws1, transposed)
    const float* __restrict__ Aj, const float* __restrict__ Bi,
    const float* __restrict__ bs1, const float* __restrict__ Ws2,
    const float* __restrict__ bs2, const float* __restrict__ mask,
    float* __restrict__ logits) {
  int bj = blockIdx.x;
  int b = bj / Sc, j = bj % Sc;
  int lane = threadIdx.x & 31;
  int wave = threadIdx.x >> 5;
  int lw = lane & 15;       // row within 16 (for A) / col (for B,C)
  int hi = lane >> 4;       // half-wave selector
  int i0 = ((int)blockIdx.y * 8 + wave) * 16;   // this wave's 16 keys

  const float* zj = Zj + (size_t)(b * Sc + j) * Dc;
  const float* zi = Zi + (size_t)(b * Sc + i0 + lw) * Dc;

  // j-invariant epilogue constants (per lane: column m = nt*16+lw)
  float ajb[6], w2v[6];
#pragma unroll
  for (int nt = 0; nt < 6; ++nt) {
    int m = nt * 16 + lw;
    ajb[nt] = Aj[(size_t)(b * Sc + j) * MHc + m] + bs1[m];
    w2v[nt] = Ws2[m];
  }
  float b2 = bs2[0];

  v8f acc[6];
#pragma unroll
  for (int nt = 0; nt < 6; ++nt) acc[nt] = (v8f){};

#pragma unroll 1
  for (int kr = 0; kr < 3; ++kr) {
    const int base = kr * 32 + hi * 8;   // index into 96-dim Z rows
    v4f zj0 = *(const v4f*)(zj + base);
    v4f zj1 = *(const v4f*)(zj + base + 4);
    v4f zj2 = *(const v4f*)(zj + base + 16);
    v4f zj3 = *(const v4f*)(zj + base + 20);
    v4f zi0 = *(const v4f*)(zi + base);
    v4f zi1 = *(const v4f*)(zi + base + 4);
    v4f zi2 = *(const v4f*)(zi + base + 16);
    v4f zi3 = *(const v4f*)(zi + base + 20);

    v16bf aP, aD;   // product fragment (kt=kr), abs-diff fragment (kt=kr+3)
#pragma unroll
    for (int e = 0; e < 4; ++e) {
      aP[e]      = (__bf16)(zj0[e] * zi0[e]);
      aP[4 + e]  = (__bf16)(zj1[e] * zi1[e]);
      aP[8 + e]  = (__bf16)(zj2[e] * zi2[e]);
      aP[12 + e] = (__bf16)(zj3[e] * zi3[e]);
      aD[e]      = (__bf16)__builtin_fabsf(zj0[e] - zi0[e]);
      aD[4 + e]  = (__bf16)__builtin_fabsf(zj1[e] - zi1[e]);
      aD[8 + e]  = (__bf16)__builtin_fabsf(zj2[e] - zi2[e]);
      aD[12 + e] = (__bf16)__builtin_fabsf(zj3[e] - zi3[e]);
    }

#pragma unroll
    for (int half = 0; half < 2; ++half) {
      const int kt = kr + 3 * half;       // 0..2 product, 3..5 absdiff
      const v16bf av = half ? aD : aP;
#pragma unroll
      for (int nt = 0; nt < 6; ++nt) {
        const __bf16* wr =
            Ws1t + (size_t)(nt * 16 + lw) * 192 + kt * 32 + hi * 8;
        v8bf b0 = *(const v8bf*)wr;
        v8bf b1 = *(const v8bf*)(wr + 16);
        v16bf bb;
#pragma unroll
        for (int e = 0; e < 8; ++e) { bb[e] = b0[e]; bb[8 + e] = b1[e]; }
        acc[nt] = wmma_bf16(av, bb, acc[nt]);
      }
    }
  }

  // epilogue: + (Aj+bs1) + Bi, relu, dot with Ws2 -> per-row logit partials
  float p[8];
#pragma unroll
  for (int r = 0; r < 8; ++r) p[r] = 0.f;

#pragma unroll
  for (int nt = 0; nt < 6; ++nt) {
    int m = nt * 16 + lw;
#pragma unroll
    for (int r = 0; r < 8; ++r) {
      int irow = i0 + r + 8 * hi;
      float h = acc[nt][r] + ajb[nt] + Bi[(size_t)(b * Sc + irow) * MHc + m];
      h = fmaxf(h, 0.f);
      p[r] += h * w2v[nt];
    }
  }
  // reduce over the 16 lanes of each half-wave (columns m within nt-tiles)
#pragma unroll
  for (int r = 0; r < 8; ++r) {
    p[r] += __shfl_xor(p[r], 1, 32);
    p[r] += __shfl_xor(p[r], 2, 32);
    p[r] += __shfl_xor(p[r], 4, 32);
    p[r] += __shfl_xor(p[r], 8, 32);
  }
  if (lw == 0) {  // lanes 0 (rows i0..i0+7) and 16 (rows i0+8..i0+15)
#pragma unroll
    for (int r = 0; r < 8; ++r) {
      int irow = i0 + r + 8 * hi;
      float lg = p[r] + b2 + (1.f - mask[b * Sc + irow]) * NEGMAX;
      logits[(size_t)(b * Sc + j) * Sc + irow] = lg;
    }
  }
}

// ------------- softmax over i + ctx = probs @ H_i + build msg_in -----------
__global__ __launch_bounds__(256) void softmax_ctx_kernel(
    const float* __restrict__ logits, const float* __restrict__ Hi,
    const float* __restrict__ Hj, __bf16* __restrict__ msgin) {
  __shared__ float sl[Sc];
  __shared__ float red[256];
  int tid = threadIdx.x;
  int b = blockIdx.x / Sc, j = blockIdx.x % Sc;
  const float* lrow = logits + (size_t)(b * Sc + j) * Sc;
  float l0 = lrow[tid], l1 = lrow[tid + 256];
  sl[tid] = l0; sl[tid + 256] = l1;
  red[tid] = fmaxf(l0, l1);
  __syncthreads();
  for (int s2 = 128; s2 > 0; s2 >>= 1) {
    if (tid < s2) red[tid] = fmaxf(red[tid], red[tid + s2]);
    __syncthreads();
  }
  float mx = red[0];
  __syncthreads();
  float e0 = __expf(l0 - mx), e1 = __expf(l1 - mx);
  sl[tid] = e0; sl[tid + 256] = e1;
  red[tid] = e0 + e1;
  __syncthreads();
  for (int s2 = 128; s2 > 0; s2 >>= 1) {
    if (tid < s2) red[tid] += red[tid + s2];
    __syncthreads();
  }
  float inv = 1.f / red[0];
  __syncthreads();

  __bf16* mrow = msgin + (size_t)(b * Sc + j) * K1c;
#pragma unroll
  for (int u = 0; u < 3; ++u) {
    int h = tid + 256 * u;
    const float* hc = Hi + (size_t)b * Sc * Hc + h;
    float a = 0.f;
    for (int i = 0; i < Sc; ++i) a += sl[i] * hc[(size_t)i * Hc];
    a *= inv;
    float hjv = Hj[(size_t)(b * Sc + j) * Hc + h];
    mrow[h]          = (__bf16)a;
    mrow[Hc + h]     = (__bf16)hjv;
    mrow[2 * Hc + h] = (__bf16)(a * hjv);
  }
}

// ---------------- generic bf16 WMMA GEMM, one 16x16 tile per wave ----------
// C = A[M,K] @ Bt[N,K]^T ; out = relu(C + bias) (bf16) or alpha*(C + bias) (f32)
__global__ __launch_bounds__(256, 4) void gemm_bf16_kernel(
    const __bf16* __restrict__ A, const __bf16* __restrict__ Bt,
    const float* __restrict__ bias, float* __restrict__ outF,
    __bf16* __restrict__ outB, int M, int N, int K, int doRelu,
    const float* __restrict__ alphaPtr) {
  int lane = threadIdx.x & 31;
  int wave = threadIdx.x >> 5;
  int tile = blockIdx.x * 8 + wave;
  int ntiles = N >> 4;
  int m0 = (tile / ntiles) * 16;
  int n0 = (tile % ntiles) * 16;
  if (m0 >= M) return;  // wave-uniform
  int lw = lane & 15, hi = lane >> 4;

  v8f acc = (v8f){};
  const __bf16* arow = A + (size_t)(m0 + lw) * K;
  const __bf16* brow = Bt + (size_t)(n0 + lw) * K;
#pragma unroll 2
  for (int kb = 0; kb < K; kb += 32) {
    const __bf16* ap = arow + kb + hi * 8;
    v8bf a0 = *(const v8bf*)ap;
    v8bf a1 = *(const v8bf*)(ap + 16);
    const __bf16* bp = brow + kb + hi * 8;
    v8bf b0 = *(const v8bf*)bp;
    v8bf b1 = *(const v8bf*)(bp + 16);
    v16bf av, bv;
#pragma unroll
    for (int e = 0; e < 8; ++e) {
      av[e] = a0[e]; av[8 + e] = a1[e];
      bv[e] = b0[e]; bv[8 + e] = b1[e];
    }
    acc = wmma_bf16(av, bv, acc);
  }
  int n = n0 + lw;
  float bb = bias[n];
  float al = alphaPtr ? *alphaPtr : 1.f;
#pragma unroll
  for (int r = 0; r < 8; ++r) {
    int row = m0 + r + 8 * hi;
    float v = acc[r] + bb;
    if (doRelu) v = fmaxf(v, 0.f);
    v *= al;
    if (outB) outB[(size_t)row * N + n] = (__bf16)v;
    else      outF[(size_t)row * N + n] = v;
  }
}

// ---------------- workspace layout (byte offsets, 256B aligned) ------------
static constexpr size_t OFF_ZJ   = 0;                              // 4*512*96 f32
static constexpr size_t OFF_ZI   = OFF_ZJ   + 786432;
static constexpr size_t OFF_AJ   = OFF_ZI   + 786432;
static constexpr size_t OFF_BI   = OFF_AJ   + 786432;
static constexpr size_t OFF_LOG  = OFF_BI   + 786432;              // 4*512*512 f32
static constexpr size_t OFF_WS1T = OFF_LOG  + 4194304;             // 96*192 bf16
static constexpr size_t OFF_WV1T = OFF_WS1T + 36864;               // 768*2304 bf16
static constexpr size_t OFF_WV2T = OFF_WV1T + 3538944;             // 768*768 bf16
static constexpr size_t OFF_MSG  = OFF_WV2T + 1179648;             // 2048*2304 bf16
static constexpr size_t OFF_Y1   = OFF_MSG  + 9437184;             // 2048*768 bf16

extern "C" void kernel_launch(void* const* d_in, const int* in_sizes, int n_in,
                              void* d_out, int out_size, void* d_ws,
                              size_t ws_size, hipStream_t stream) {
  const float* H_j  = (const float*)d_in[0];
  const float* H_i  = (const float*)d_in[1];
  const float* mask = (const float*)d_in[2];
  const float* Wpj  = (const float*)d_in[3];
  const float* Wpi  = (const float*)d_in[4];
  const float* Ws1  = (const float*)d_in[5];
  const float* bs1  = (const float*)d_in[6];
  const float* Ws2  = (const float*)d_in[7];
  const float* bs2  = (const float*)d_in[8];
  const float* Wv1  = (const float*)d_in[9];
  const float* bv1  = (const float*)d_in[10];
  const float* Wv2  = (const float*)d_in[11];
  const float* bv2  = (const float*)d_in[12];
  const float* alpha = (const float*)d_in[13];

  char* ws = (char*)d_ws;
  float*  Zj     = (float*)(ws + OFF_ZJ);
  float*  Zi     = (float*)(ws + OFF_ZI);
  float*  Aj     = (float*)(ws + OFF_AJ);
  float*  Bi     = (float*)(ws + OFF_BI);
  float*  logits = (float*)(ws + OFF_LOG);
  __bf16* Ws1t   = (__bf16*)(ws + OFF_WS1T);
  __bf16* Wv1t   = (__bf16*)(ws + OFF_WV1T);
  __bf16* Wv2t   = (__bf16*)(ws + OFF_WV2T);
  __bf16* msgin  = (__bf16*)(ws + OFF_MSG);
  __bf16* Y1     = (__bf16*)(ws + OFF_Y1);

  const int rows = Bc * Sc;  // 2048

  // 1) weight prep (transpose + bf16)
  transpose_bf16_kernel<<<(192 * 96 + 255) / 256, 256, 0, stream>>>(
      Ws1 + (size_t)2 * Dc * MHc, Ws1t, 192, 96);   // rows 192..383 of Ws1
  transpose_bf16_kernel<<<(K1c * OHc + 255) / 256, 256, 0, stream>>>(
      Wv1, Wv1t, K1c, OHc);
  transpose_bf16_kernel<<<(OHc * Hc + 255) / 256, 256, 0, stream>>>(
      Wv2, Wv2t, OHc, Hc);

  // 2) projections, 3) separable score terms
  proj_kernel<<<rows, 256, 0, stream>>>(H_j, H_i, Wpj, Wpi, Zj, Zi);
  ajbi_kernel<<<rows, 256, 0, stream>>>(Zj, Zi, Ws1, Aj, Bi);

  // 4) pair-score logits (WMMA): grid (B*S, 4), one i-tile per wave
  pair_kernel<<<dim3(rows, 4), 256, 0, stream>>>(Zj, Zi, Ws1t, Aj, Bi, bs1,
                                                 Ws2, bs2, mask, logits);

  // 5) softmax + context + msg_in
  softmax_ctx_kernel<<<rows, 256, 0, stream>>>(logits, H_i, H_j, msgin);

  // 6) Y1 = relu(msg_in @ Wv1 + bv1)   M=2048 N=768 K=2304
  {
    int tiles = (rows / 16) * (OHc / 16);  // 6144
    gemm_bf16_kernel<<<tiles / 8, 256, 0, stream>>>(
        msgin, Wv1t, bv1, nullptr, Y1, rows, OHc, K1c, 1, nullptr);
  }
  // 7) out = alpha * (Y1 @ Wv2 + bv2)  M=2048 N=768 K=768
  {
    int tiles = (rows / 16) * (Hc / 16);   // 6144
    gemm_bf16_kernel<<<tiles / 8, 256, 0, stream>>>(
        Y1, Wv2t, bv2, (float*)d_out, nullptr, rows, Hc, OHc, 0, alpha);
  }
}
